// GraphTransformerNet_52948356825798
// MI455X (gfx1250) — matrix operand order repaired and checked by reading, not compiled, not run on hardware
//
#include <hip/hip_runtime.h>
#include <stdint.h>

// ---------------- types for WMMA ----------------
typedef __attribute__((ext_vector_type(16))) __bf16 v16bf;
typedef __attribute__((ext_vector_type(8)))  float  v8f;

union Frag { v16bf v; uint4 q[2]; };

#define AW  520   // A row stride in bf16 elems (512 + 8 pad)
#define KVW 264   // k/v row stride in bf16 elems (256 + 8 pad)

static constexpr int Bn = 1024, Nn = 128, Dn = 256, HC = 256;

__device__ __forceinline__ uint16_t f2bf(float f) {
  uint32_t u = __float_as_uint(f);
  return (uint16_t)((u + 0x7FFFu + ((u >> 16) & 1u)) >> 16);  // RNE
}
__device__ __forceinline__ float bf2f(uint16_t h) {
  return __uint_as_float(((uint32_t)h) << 16);
}

// ---------------------------------------------------------------------------
// Pack kernel: build bf16 B-matrix fragments in d_ws, tile layout:
//   tiles [0,256):   Wkcat = [Wk ; We]   (K=512, 16 kt x 16 nt)
//   tiles [256,512): Wvcat = [Wv ; We]
//   tiles [512,640): Wskip              (K=256,  8 kt x 16 nt)
//   tiles [640,768): Wq
// Each tile = 512 bf16 = 256 u32; lane l owns u32 [l*8 .. l*8+7] so the GEMM
// kernel fetches a whole 32x16 B tile with two global_load_b128 per lane.
// B VGPR layout: lanes 0-15 K=0..15 (VGPR j: K=2j,2j+1), lanes 16-31 K=16..31,
// n = lane&15  (contiguous k order, same as the A loads).
// ---------------------------------------------------------------------------
__global__ void pack_weights(const float* __restrict__ Wq, const float* __restrict__ Wk,
                             const float* __restrict__ Wv, const float* __restrict__ We,
                             const float* __restrict__ Wsk, uint32_t* __restrict__ ws) {
  int tile = blockIdx.x;
  int t = threadIdx.x;
  int l = t >> 3, j = t & 7;

  const float* W; int kt, nt; bool cat = false;
  if (tile < 256)      { int tt = tile;       kt = tt >> 4; nt = tt & 15; W = Wk;  cat = true; }
  else if (tile < 512) { int tt = tile - 256; kt = tt >> 4; nt = tt & 15; W = Wv;  cat = true; }
  else if (tile < 640) { int tt = tile - 512; kt = tt >> 4; nt = tt & 15; W = Wsk; }
  else                 { int tt = tile - 640; kt = tt >> 4; nt = tt & 15; W = Wq;  }

  int krow = kt * 32 + ((l >> 4) << 4) + 2 * j;   // even; pair never straddles 32-block
  int col  = nt * 16 + (l & 15);
  const float* src = W;
  int kr = krow;
  if (cat && krow >= 256) { src = We; kr = krow - 256; }
  float f0 = src[(size_t)kr * HC + col];
  float f1 = src[(size_t)(kr + 1) * HC + col];
  ws[(size_t)tile * 256 + t] = (uint32_t)f2bf(f0) | ((uint32_t)f2bf(f1) << 16);
}

// ---------------------------------------------------------------------------
// Fused graph-transformer kernel: one workgroup per graph.
// ---------------------------------------------------------------------------
extern "C" __global__ void __launch_bounds__(256)
graph_tconv(const float* __restrict__ central, const float* __restrict__ nbr,
            const float* __restrict__ edge,
            const float* __restrict__ bq, const float* __restrict__ bk,
            const float* __restrict__ bv, const float* __restrict__ bsk,
            const uint32_t* __restrict__ wsu, float* __restrict__ out) {
  extern __shared__ char smem[];
  uint16_t* A    = (uint16_t*)smem;                       // 144 x AW  bf16
  uint16_t* Kt   = (uint16_t*)(smem + 144 * AW * 2);      // 128 x KVW bf16
  uint16_t* Vt   = Kt + 128 * KVW;                        // 128 x KVW bf16
  float*    sc   = (float*)(Vt + 128 * KVW);              // 4 x 128 scores/alpha
  float*    qv   = sc + 512;                              // 256 q row
  float*    csk  = qv + 256;                              // 256 central skip row
  float*    bias = csk + 256;                             // [bq|bk|bv|bskip] 4x256

  const int b    = blockIdx.x;
  const int t    = threadIdx.x;
  const int lane = t & 31;
  const int wv   = t >> 5;

  // ---- stage biases ----
  bias[t]       = bq[t];
  bias[256 + t] = bk[t];
  bias[512 + t] = bv[t];
  bias[768 + t] = bsk[t];

  // ---- stage A = [nbr | edge] rows 0..127, [central | 0] row 128, zeros 129..143 ----
  const float* nb = nbr  + (size_t)b * Nn * Dn;
  const float* eg = edge + (size_t)b * Nn * Dn;
#pragma unroll 4
  for (int i = 0; i < 64; ++i) {
    int chunk = t + 256 * i;                 // 16384 float4-chunks total
    int row   = chunk >> 7;                  // 128 chunks per 512-wide row
    int cp    = (chunk & 127) * 4;
    const float4 f = *(const float4*)((cp < 256) ? (nb + (size_t)row * Dn + cp)
                                                 : (eg + (size_t)row * Dn + cp - 256));
    uint32_t lo = (uint32_t)f2bf(f.x) | ((uint32_t)f2bf(f.y) << 16);
    uint32_t hi = (uint32_t)f2bf(f.z) | ((uint32_t)f2bf(f.w) << 16);
    *(uint2*)&A[row * AW + cp] = make_uint2(lo, hi);
  }
  if (t < 64) {
    const float4 f = *(const float4*)(central + (size_t)b * Dn + t * 4);
    uint32_t lo = (uint32_t)f2bf(f.x) | ((uint32_t)f2bf(f.y) << 16);
    uint32_t hi = (uint32_t)f2bf(f.z) | ((uint32_t)f2bf(f.w) << 16);
    *(uint2*)&A[128 * AW + t * 4] = make_uint2(lo, hi);
  } else if (t < 128) {
    *(uint2*)&A[128 * AW + 256 + (t - 64) * 4] = make_uint2(0u, 0u);
  }
  for (int idx = t; idx < 15 * AW; idx += 256) A[129 * AW + idx] = 0;
  __syncthreads();

  const uint32_t* Bk  = wsu;              // Wkcat
  const uint32_t* Bv  = wsu + 256 * 256;  // Wvcat
  const uint32_t* Bs  = wsu + 512 * 256;  // Wskip
  const uint32_t* Bqw = wsu + 640 * 256;  // Wq

  // One 16-row M strip x 8 N tiles, K accumulated over nkt 32-wide steps.
  auto run_gemm = [&](int mtile, int nkt, const uint32_t* Bt, v8f* acc, int half) {
    const v8f zero = {0.f, 0.f, 0.f, 0.f, 0.f, 0.f, 0.f, 0.f};
#pragma unroll
    for (int i = 0; i < 8; ++i) acc[i] = zero;
    const int row = mtile * 16 + (lane & 15);
    const int khi = (lane >> 4) * 16;           // A: lanes 0-15 K 0..15, 16-31 K 16..31
    const uint4* Bq4 = (const uint4*)Bt;
    for (int kt = 0; kt < nkt; ++kt) {
      Frag a;
      const uint16_t* pa = &A[row * AW + kt * 32 + khi];
      a.q[0] = *(const uint4*)pa;               // K = khi .. khi+7
      a.q[1] = *(const uint4*)(pa + 8);         // K = khi+8 .. khi+15
#pragma unroll
      for (int n8 = 0; n8 < 8; ++n8) {
        Frag bf;
        const int tile = kt * 16 + half * 8 + n8;
        bf.q[0] = Bq4[tile * 64 + lane * 2];
        bf.q[1] = Bq4[tile * 64 + lane * 2 + 1];
        acc[n8] = __builtin_amdgcn_wmma_f32_16x16x32_bf16(
            false, a.v, false, bf.v, (short)0, acc[n8], false, false);
      }
    }
  };

  v8f acc[8];
  const int mbase = wv * 16 + 8 * (lane >> 4);  // D layout: VGPR r -> M = r (+8 hi half)

  // ---- k and v: K=512 over [nbr|edge], bf16 results into LDS ----
  for (int half = 0; half < 2; ++half) {
    run_gemm(wv, 16, Bk, acc, half);
#pragma unroll
    for (int n8 = 0; n8 < 8; ++n8) {
      const int col = (half * 8 + n8) * 16 + (lane & 15);
      const float bb = bias[256 + col];
#pragma unroll
      for (int r = 0; r < 8; ++r) Kt[(mbase + r) * KVW + col] = f2bf(acc[n8][r] + bb);
    }
    run_gemm(wv, 16, Bv, acc, half);
#pragma unroll
    for (int n8 = 0; n8 < 8; ++n8) {
      const int col = (half * 8 + n8) * 16 + (lane & 15);
      const float bb = bias[512 + col];
#pragma unroll
      for (int r = 0; r < 8; ++r) Vt[(mbase + r) * KVW + col] = f2bf(acc[n8][r] + bb);
    }
  }

  // ---- neighbor skip rows: K=256 over nbr half of A, straight to global ----
  for (int half = 0; half < 2; ++half) {
    run_gemm(wv, 8, Bs, acc, half);
#pragma unroll
    for (int n8 = 0; n8 < 8; ++n8) {
      const int col = (half * 8 + n8) * 16 + (lane & 15);
      const float bb = bias[768 + col];
#pragma unroll
      for (int r = 0; r < 8; ++r)
        out[((size_t)b * 129 + 1 + mbase + r) * HC + col] = acc[n8][r] + bb;
    }
  }

  // ---- central row: skip (wave 0) and q (wave 1) via m-tile 8 (row 128 valid) ----
  if (wv == 0) {
    for (int half = 0; half < 2; ++half) {
      run_gemm(8, 8, Bs, acc, half);
      if (lane < 16)
#pragma unroll
        for (int n8 = 0; n8 < 8; ++n8) {
          const int col = (half * 8 + n8) * 16 + lane;
          csk[col] = acc[n8][0] + bias[768 + col];
        }
    }
  } else if (wv == 1) {
    for (int half = 0; half < 2; ++half) {
      run_gemm(8, 8, Bqw, acc, half);
      if (lane < 16)
#pragma unroll
        for (int n8 = 0; n8 < 8; ++n8) {
          const int col = (half * 8 + n8) * 16 + lane;
          qv[col] = acc[n8][0] + bias[col];
        }
    }
  }
  __syncthreads();

  // ---- attention scores: s[h][n] = (q_h . k_n) / sqrt(64) ----
  for (int p = t; p < 512; p += 256) {
    const int h = p >> 7, n = p & 127;
    const uint16_t* kr = &Kt[n * KVW + h * 64];
    const float* qh = &qv[h * 64];
    float s = 0.f;
#pragma unroll 8
    for (int c = 0; c < 64; ++c) s += qh[c] * bf2f(kr[c]);
    sc[h * 128 + n] = s * 0.125f;
  }
  __syncthreads();

  // ---- softmax over the 128 neighbors: one wave32 per head ----
  if (wv < 4) {
    const int h = wv;
    float x0 = sc[h * 128 + lane],      x1 = sc[h * 128 + lane + 32];
    float x2 = sc[h * 128 + lane + 64], x3 = sc[h * 128 + lane + 96];
    float m = fmaxf(fmaxf(x0, x1), fmaxf(x2, x3));
#pragma unroll
    for (int o = 16; o >= 1; o >>= 1) m = fmaxf(m, __shfl_xor(m, o, 32));
    float e0 = __expf(x0 - m), e1 = __expf(x1 - m), e2 = __expf(x2 - m), e3 = __expf(x3 - m);
    float ssum = e0 + e1 + e2 + e3;
#pragma unroll
    for (int o = 16; o >= 1; o >>= 1) ssum += __shfl_xor(ssum, o, 32);
    const float inv = 1.f / ssum;
    sc[h * 128 + lane]      = e0 * inv; sc[h * 128 + lane + 32] = e1 * inv;
    sc[h * 128 + lane + 64] = e2 * inv; sc[h * 128 + lane + 96] = e3 * inv;
  }
  __syncthreads();

  // ---- aggregate messages + central output row ----
  {
    const int h = t >> 6, c = t & 63;
    const float* al = &sc[h * 128];
    float s = 0.f;
#pragma unroll 8
    for (int n = 0; n < 128; ++n) s += al[n] * bf2f(Vt[n * KVW + h * 64 + c]);
    out[(size_t)b * 129 * HC + t] = csk[t] + s;
  }
}

// ---------------------------------------------------------------------------
extern "C" void kernel_launch(void* const* d_in, const int* in_sizes, int n_in,
                              void* d_out, int out_size, void* d_ws, size_t ws_size,
                              hipStream_t stream) {
  (void)in_sizes; (void)n_in; (void)out_size; (void)ws_size;
  const float* central = (const float*)d_in[0];
  const float* nbr     = (const float*)d_in[1];
  const float* edge    = (const float*)d_in[2];
  const float* Wq  = (const float*)d_in[3];
  const float* bq  = (const float*)d_in[4];
  const float* Wk  = (const float*)d_in[5];
  const float* bk  = (const float*)d_in[6];
  const float* Wv  = (const float*)d_in[7];
  const float* bv  = (const float*)d_in[8];
  const float* We  = (const float*)d_in[9];
  const float* Wsk = (const float*)d_in[10];
  const float* bsk = (const float*)d_in[11];
  uint32_t* ws = (uint32_t*)d_ws;

  pack_weights<<<768, 256, 0, stream>>>(Wq, Wk, Wv, We, Wsk, ws);

  const size_t lds_bytes = (size_t)(144 * AW + 2 * 128 * KVW) * 2  // A, k, v (bf16)
                         + (512 + 256 + 256 + 1024) * 4;           // scores,q,csk,bias
  (void)hipFuncSetAttribute((const void*)graph_tconv,
                            hipFuncAttributeMaxDynamicSharedMemorySize,
                            (int)lds_bytes);
  graph_tconv<<<1024, 256, lds_bytes, stream>>>(central, nbr, edge,
                                                bq, bk, bv, bsk, ws, (float*)d_out);
}
